// GRU_25546465477224
// MI455X (gfx1250) — matrix operand order repaired and checked by reading, not compiled, not run on hardware
//
#include <hip/hip_runtime.h>
#include <hip/hip_bf16.h>

// GRU scan on gfx1250: one wave32 per 16-row batch tile, bf16 WMMA (f32 acc),
// weights register-resident in B-layout, h kept in C-layout registers,
// LDS used only for the cross-lane C->A transpose of h / (h*r).
// Software pipeline: x-index prefetched 2 steps ahead, embedding rows 1 step
// ahead. In-loop barriers replaced by sched_barrier: same-wave LDS ops are
// in-order in hardware (DScnt), so only compiler reordering must be fenced.

typedef __attribute__((ext_vector_type(16))) __bf16 v16bf;
typedef __attribute__((ext_vector_type(8)))  float  v8f;

#define TT   2048
#define BB   2048
#define EMBD 64
#define LAT  32
#define KD   96   // EMBD + LAT

static __device__ inline v8f wmma_bf16(v16bf a, v16bf b, v8f c) {
  // v_wmma_f32_16x16x32_bf16  D = A(16x32) * B(32x16) + C
  return __builtin_amdgcn_wmma_f32_16x16x32_bf16(
      /*neg_a=*/false, a, /*neg_b=*/false, b,
      /*c_mod=*/(short)0, c, /*reuse_a=*/false, /*reuse_b=*/false);
}

// Pack 16 consecutive f32 (as 4 float4) into a bf16 fragment, element j = src[j].
static __device__ inline v16bf pack16(float4 a, float4 b, float4 c, float4 d) {
  v16bf v;
  v[0]  = (__bf16)a.x; v[1]  = (__bf16)a.y; v[2]  = (__bf16)a.z; v[3]  = (__bf16)a.w;
  v[4]  = (__bf16)b.x; v[5]  = (__bf16)b.y; v[6]  = (__bf16)b.z; v[7]  = (__bf16)b.w;
  v[8]  = (__bf16)c.x; v[9]  = (__bf16)c.y; v[10] = (__bf16)c.z; v[11] = (__bf16)c.w;
  v[12] = (__bf16)d.x; v[13] = (__bf16)d.y; v[14] = (__bf16)d.z; v[15] = (__bf16)d.w;
  return v;
}

static __device__ inline v8f splat8(float s) {
  v8f v;
#pragma unroll
  for (int i = 0; i < 8; ++i) v[i] = s;
  return v;
}

static __device__ inline v8f sigmoid8(v8f x) {
  v8f y;
#pragma unroll
  for (int i = 0; i < 8; ++i) {
    float e = __builtin_amdgcn_exp2f(x[i] * -1.44269504f);   // e^{-x}
    y[i] = __builtin_amdgcn_rcpf(1.0f + e);
  }
  return y;
}

static __device__ inline v8f tanh8(v8f x) {
  v8f y;
#pragma unroll
  for (int i = 0; i < 8; ++i) {
    float e = __builtin_amdgcn_exp2f(x[i] * 2.885390082f);   // e^{2x}
    y[i] = 1.0f - 2.0f * __builtin_amdgcn_rcpf(1.0f + e);    // tanh = 1 - 2/(e^{2x}+1)
  }
  return y;
}

// A-layout fragment for one 16x32 K-chunk from LDS, two 8-float runs per lane:
// run0 at K0 + hi*8, run1 at K0 + 16 + hi*8   (rows striped: lane&15 = row).
static __device__ inline v16bf load_A_chunk_lds(const float* rowp, int hi) {
  const float4* p = (const float4*)(rowp + hi * 8);
  float4 s0 = p[0], s1 = p[1];      // K0 + hi*8      .. +7
  float4 s2 = p[4], s3 = p[5];      // K0 + 16 + hi*8 .. +7
  return pack16(s0, s1, s2, s3);
}

__global__ __launch_bounds__(32)
void gru_wmma_kernel(const int*   __restrict__ x,
                     const float* __restrict__ h0,
                     const float* __restrict__ emb,
                     const float* __restrict__ Wz, const float* __restrict__ bz,
                     const float* __restrict__ Wr, const float* __restrict__ br,
                     const float* __restrict__ Wh, const float* __restrict__ bh,
                     float*       __restrict__ out) {
  __shared__ float h_lds[16 * LAT];   // current h, row-major [16][32]
  __shared__ float s_lds[16 * LAT];   // scratch for h*r transpose

  const int lane = threadIdx.x;
  const int n    = lane & 15;        // C-layout column / A-layout row / B-layout column
  const int hi   = lane >> 4;        // lane half
  const int base = blockIdx.x * 16;  // batch-tile base row

  // ---------- weight fragments, B-layout, bf16, register-resident ----------
  // B(32x16) per (gate, n-tile, k-chunk): lane col = n, elements 0..7 hold K = hi*8 + s, 8..15 hold K = 16 + hi*8 + (s-8)
  v16bf BW[3][2][3];
  {
    const float* Wp[3] = {Wz, Wr, Wh};
#pragma unroll
    for (int g = 0; g < 3; ++g)
#pragma unroll
      for (int nt = 0; nt < 2; ++nt)
#pragma unroll
        for (int c = 0; c < 3; ++c) {
          const float4* p = (const float4*)(Wp[g] + (size_t)(nt * 16 + n) * KD + c * 32 + hi * 8);
          BW[g][nt][c] = pack16(p[0], p[1], p[4], p[5]);
        }
  }

  // Loop-invariant bias seed vectors (C operand of the first K-chunk WMMA).
  const v8f cz0 = splat8(bz[n]), cz1 = splat8(bz[16 + n]);
  const v8f cr0 = splat8(br[n]), cr1 = splat8(br[16 + n]);
  const v8f cq0 = splat8(bh[n]), cq1 = splat8(bh[16 + n]);

  // ---------- initial h in C-layout registers + LDS mirror ----------
  v8f hC0, hC1;
#pragma unroll
  for (int r = 0; r < 8; ++r) {
    int m = r + 8 * hi;
    hC0[r] = h0[(size_t)(base + m) * LAT + n];
    hC1[r] = h0[(size_t)(base + m) * LAT + 16 + n];
  }
#pragma unroll
  for (int r = 0; r < 8; ++r) {
    int m = r + 8 * hi;
    h_lds[m * LAT + n]      = hC0[r];
    h_lds[m * LAT + 16 + n] = hC1[r];
  }
  __syncthreads();

  // ---------- software-pipeline prologue ----------
  v16bf a0, a1;     // bf16 A-fragments for the *current* step's embedding
  {
    int id0 = x[(size_t)0 * BB + base + n];
    const float4* ep = (const float4*)(emb + (size_t)id0 * EMBD + hi * 8);
    a0 = pack16(ep[0], ep[1], ep[4],  ep[5]);
    a1 = pack16(ep[8], ep[9], ep[12], ep[13]);
  }
  int id_b = x[(size_t)1 * BB + base + n];   // token id for step t+1

  for (int t = 0; t < TT; ++t) {
    // ---- prefetch token id for t+2 (uniform row pointer -> saddr addressing)
    size_t tc = (size_t)((t + 2 < TT) ? (t + 2) : (TT - 1));
    const int* xrow = x + tc * BB + base;
    int id_c = xrow[n];

    // ---- issue embedding loads for step t+1 (id_b already resident)
    const float4* epn = (const float4*)(emb + (size_t)id_b * EMBD + hi * 8);
    float4 f0 = epn[0], f1 = epn[1], f2 = epn[4],  f3 = epn[5];
    float4 f4 = epn[8], f5 = epn[9], f6 = epn[12], f7 = epn[13];

    // ---- h-independent partial sums: bias-seeded + two embedding K-chunks
    v8f z0 = wmma_bf16(a0, BW[0][0][0], cz0);  z0 = wmma_bf16(a1, BW[0][0][1], z0);
    v8f z1 = wmma_bf16(a0, BW[0][1][0], cz1);  z1 = wmma_bf16(a1, BW[0][1][1], z1);
    v8f r0 = wmma_bf16(a0, BW[1][0][0], cr0);  r0 = wmma_bf16(a1, BW[1][0][1], r0);
    v8f r1 = wmma_bf16(a0, BW[1][1][0], cr1);  r1 = wmma_bf16(a1, BW[1][1][1], r1);
    v8f q0 = wmma_bf16(a0, BW[2][0][0], cq0);  q0 = wmma_bf16(a1, BW[2][0][1], q0);
    v8f q1 = wmma_bf16(a0, BW[2][1][0], cq1);  q1 = wmma_bf16(a1, BW[2][1][1], q1);

    // ---- a0/a1 now dead: repack with step-(t+1) data (loads have had cover)
    a0 = pack16(f0, f1, f2, f3);
    a1 = pack16(f4, f5, f6, f7);
    id_b = id_c;

    // ---- serial chain: h -> A-layout (via LDS); r first (needed for h*r)
    v16bf aH = load_A_chunk_lds(&h_lds[n * LAT], hi);
    r0 = wmma_bf16(aH, BW[1][0][2], r0);  r1 = wmma_bf16(aH, BW[1][1][2], r1);
    z0 = wmma_bf16(aH, BW[0][0][2], z0);  z1 = wmma_bf16(aH, BW[0][1][2], z1);
    r0 = sigmoid8(r0);  r1 = sigmoid8(r1);

    // h * r -> LDS -> A-layout  (same-wave LDS is in-order; fence the compiler)
    v8f hr0 = hC0 * r0, hr1 = hC1 * r1;
#pragma unroll
    for (int r = 0; r < 8; ++r) {
      int m = r + 8 * hi;
      s_lds[m * LAT + n]      = hr0[r];
      s_lds[m * LAT + 16 + n] = hr1[r];
    }
    __builtin_amdgcn_sched_barrier(0);

    // z-sigmoid executes in the shadow of the s_lds->aR->WMMA->tanh chain
    z0 = sigmoid8(z0);  z1 = sigmoid8(z1);

    v16bf aR = load_A_chunk_lds(&s_lds[n * LAT], hi);
    q0 = wmma_bf16(aR, BW[2][0][2], q0);
    q1 = wmma_bf16(aR, BW[2][1][2], q1);
    q0 = tanh8(q0);  q1 = tanh8(q1);

    // h_new = h + z*(h~ - h)
    hC0 = hC0 + z0 * (q0 - hC0);
    hC1 = hC1 + z1 * (q1 - hC1);

    // publish h for next step + write output tile (contiguous 2KB span)
    __builtin_amdgcn_sched_barrier(0);
#pragma unroll
    for (int r = 0; r < 8; ++r) {
      int m = r + 8 * hi;
      h_lds[m * LAT + n]      = hC0[r];
      h_lds[m * LAT + 16 + n] = hC1[r];
    }
    float* orow = out + ((size_t)t * BB + base) * LAT;
#pragma unroll
    for (int r = 0; r < 8; ++r) {
      int m = r + 8 * hi;
      orow[m * LAT + n]      = hC0[r];
      orow[m * LAT + 16 + n] = hC1[r];
    }
    __builtin_amdgcn_sched_barrier(0);
  }
}

extern "C" void kernel_launch(void* const* d_in, const int* in_sizes, int n_in,
                              void* d_out, int out_size, void* d_ws, size_t ws_size,
                              hipStream_t stream) {
  (void)in_sizes; (void)n_in; (void)out_size; (void)d_ws; (void)ws_size;
  const int*   x   = (const int*)  d_in[0];
  const float* h0  = (const float*)d_in[1];
  const float* emb = (const float*)d_in[2];
  const float* Wz  = (const float*)d_in[3];
  const float* bz  = (const float*)d_in[4];
  const float* Wr  = (const float*)d_in[5];
  const float* br  = (const float*)d_in[6];
  const float* Wh  = (const float*)d_in[7];
  const float* bh  = (const float*)d_in[8];
  float* out = (float*)d_out;

  dim3 grid(BB / 16), block(32);
  hipLaunchKernelGGL(gru_wmma_kernel, grid, block, 0, stream,
                     x, h0, emb, Wz, bz, Wr, br, Wh, bh, out);
}
